// MHAEncoderFusedProj_42167988912449
// MI455X (gfx1250) — matrix operand order repaired and checked by reading, not compiled, not run on hardware
//
#include <hip/hip_runtime.h>
#include <hip/hip_bf16.h>

// ---------------------------------------------------------------------------
// MHA encoder (B=4, S=2048, D=1024, H=16, DH=64) for gfx1250 (CDNA5, wave32)
// Matmuls on v_wmma_f32_16x16x32_f16; tile staging via CDNA5 async
// global->LDS copies (global_load_async_to_lds_b128, ASYNCcnt) with
// double-buffered LDS so loads overlap WMMA compute.
// ---------------------------------------------------------------------------

typedef __attribute__((ext_vector_type(16))) _Float16 v16h;
typedef __attribute__((ext_vector_type(8)))  _Float16 v8h;
typedef __attribute__((ext_vector_type(8)))  float    v8f;

union Frag16 { v16h v; v8h h[2]; };

#define B_   4
#define S_   2048
#define D_   1024
#define H_   16
#define DH_  64
#define M_   (B_ * S_)    // 8192 rows
#define N3_  (3 * D_)     // 3072 qkv cols

// Async global->LDS 16B copy, GVS mode: SGPR 64-bit base + per-lane u32 offset.
// lds = wave-relative LDS byte address (low 32 bits of a __shared__ pointer).
#define ASYNC_CP16(lds, off, base)                                            \
    asm volatile("global_load_async_to_lds_b128 %0, %1, %2"                   \
                 :: "v"(lds), "v"(off), "s"(base) : "memory")

__device__ __forceinline__ uint32_t lds_addr(const void* p) {
    return (uint32_t)(uintptr_t)p;   // generic LDS pointer: low 32 bits = LDS offset
}

// ---------------------------------------------------------------------------
// fp32 -> f16 conversion
// ---------------------------------------------------------------------------
__global__ __launch_bounds__(256) void cvt_f32_f16(const float* __restrict__ src,
                                                   _Float16* __restrict__ dst, int n) {
    int i = blockIdx.x * 256 + threadIdx.x;
    if (i < n) dst[i] = (_Float16)src[i];
}

// ---------------------------------------------------------------------------
// Tiled GEMM: C[M,N] = A[M,K] * Wt[N,K]^T  (Wt row-major [N][K] = WMMA B layout)
// BM=128, BN=64, BK=64; 4 waves, each wave owns 32 rows x 64 cols.
// Double-buffered LDS, async global->LDS staging.
// MODE 0: scatter into q / k / v^T f16 buffers   MODE 1: fp32 out
// ---------------------------------------------------------------------------
template <int MODE>
__global__ __launch_bounds__(128) void gemm_wmma(
    const _Float16* __restrict__ A, const _Float16* __restrict__ Wt,
    int Kdim, int Ndim,
    _Float16* __restrict__ qo, _Float16* __restrict__ ko_,
    _Float16* __restrict__ vto, float* __restrict__ outf)
{
    __shared__ __align__(16) _Float16 As[2][128][72];   // +8 pad vs bank conflicts
    __shared__ __align__(16) _Float16 Bs[2][64][72];

    const int tid  = threadIdx.x;
    const int wave = tid >> 5, lane = tid & 31;
    const int g = lane >> 4, ln = lane & 15;
    const int bm = blockIdx.x * 128, bn = blockIdx.y * 64;
    const int wr = wave * 32;

    // Issue one 64-wide K chunk into LDS buffer `buf` (10 async instr / thread).
    auto issue = [&](int buf, int k0) {
        // A tile: thread tid copies its row (8 x 16B)
        uint32_t ldsA = lds_addr(&As[buf][tid][0]);
        uint32_t offA = (uint32_t)((bm + tid) * Kdim + k0) * 2u;
        #pragma unroll
        for (int i = 0; i < 8; ++i)
            ASYNC_CP16(ldsA + i * 16, offA + i * 16, A);
        // B tile: 2 threads per row, 2 x 16B each
        int row = tid >> 1, seg = tid & 1;
        uint32_t ldsB = lds_addr(&Bs[buf][row][seg * 32]);
        uint32_t offB = (uint32_t)((bn + row) * Kdim + k0 + seg * 32) * 2u;
        ASYNC_CP16(ldsB,      offB,      Wt);
        ASYNC_CP16(ldsB + 16, offB + 16, Wt);
    };

    v8f acc[2][4];
    for (int mt = 0; mt < 2; ++mt)
        for (int nt = 0; nt < 4; ++nt)
            for (int r = 0; r < 8; ++r) acc[mt][nt][r] = 0.0f;

    const int nk = Kdim >> 6;
    issue(0, 0);
    for (int ic = 0; ic < nk; ++ic) {
        const int cur = ic & 1;
        if (ic + 1 < nk) {
            issue(cur ^ 1, (ic + 1) * 64);   // prefetch next chunk (overlaps compute)
            asm volatile("s_wait_asynccnt 0xa" ::: "memory");  // cur chunk landed
        } else {
            asm volatile("s_wait_asynccnt 0x0" ::: "memory");
        }
        __syncthreads();   // all waves' cur-chunk data visible

        #pragma unroll
        for (int ko2 = 0; ko2 < 2; ++ko2) {              // two K=32 steps
            Frag16 a0, a1;                                // A-frag: k = ko + g*8 (+16)
            a0.h[0] = *(const v8h*)&As[cur][wr + ln     ][ko2 * 32 + g * 8];
            a0.h[1] = *(const v8h*)&As[cur][wr + ln     ][ko2 * 32 + 16 + g * 8];
            a1.h[0] = *(const v8h*)&As[cur][wr + 16 + ln][ko2 * 32 + g * 8];
            a1.h[1] = *(const v8h*)&As[cur][wr + 16 + ln][ko2 * 32 + 16 + g * 8];
            #pragma unroll
            for (int nt = 0; nt < 4; ++nt) {
                Frag16 bb;                                // B-frag: k = ko + g*8 (+16)
                bb.h[0] = *(const v8h*)&Bs[cur][nt * 16 + ln][ko2 * 32 + g * 8];
                bb.h[1] = *(const v8h*)&Bs[cur][nt * 16 + ln][ko2 * 32 + 16 + g * 8];
                acc[0][nt] = __builtin_amdgcn_wmma_f32_16x16x32_f16(
                    false, a0.v, false, bb.v, (short)0, acc[0][nt], false, false);
                acc[1][nt] = __builtin_amdgcn_wmma_f32_16x16x32_f16(
                    false, a1.v, false, bb.v, (short)0, acc[1][nt], false, false);
            }
        }
        __syncthreads();   // readers done before next iter re-issues this buffer
    }

    // Epilogue. C layout: row = g*8 + r, col = ln (per 16x16 tile).
    #pragma unroll
    for (int mt = 0; mt < 2; ++mt) {
        #pragma unroll
        for (int nt = 0; nt < 4; ++nt) {
            #pragma unroll
            for (int r = 0; r < 8; ++r) {
                int row = bm + wr + mt * 16 + g * 8 + r;
                int col = bn + nt * 16 + ln;
                float val = acc[mt][nt][r];
                if constexpr (MODE == 0) {
                    int t = col >> 10, hh = (col >> 6) & 15, dh = col & 63;
                    int bb = row >> 11, s = row & (S_ - 1);
                    size_t bh = (size_t)bb * H_ + hh;
                    if (t == 0)      qo [(bh * S_ + s) * DH_ + dh]  = (_Float16)val;
                    else if (t == 1) ko_[(bh * S_ + s) * DH_ + dh]  = (_Float16)val;
                    else             vto[(bh * DH_ + dh) * S_ + s]  = (_Float16)val; // V^T
                } else {
                    outf[(size_t)row * Ndim + col] = val;
                }
            }
        }
    }
}

// ---------------------------------------------------------------------------
// RoPE (rotate-half) on q and k, fp32 math, in place; folds 1/sqrt(DH) into q.
// ---------------------------------------------------------------------------
__global__ __launch_bounds__(256) void rope_kernel(_Float16* __restrict__ q,
                                                   _Float16* __restrict__ k,
                                                   const float* __restrict__ cp,
                                                   const float* __restrict__ sp)
{
    int i  = blockIdx.x * 256 + threadIdx.x;       // i < B*H*S*32
    int dh = i & 31;
    int s  = (i >> 5) & (S_ - 1);
    int bh = i >> 16;                               // B*H = 64
    size_t base = ((size_t)bh * S_ + s) * DH_;
    float c0 = cp[s * DH_ + dh],      s0 = sp[s * DH_ + dh];
    float c1 = cp[s * DH_ + dh + 32], s1 = sp[s * DH_ + dh + 32];

    float q0 = (float)q[base + dh], q1 = (float)q[base + dh + 32];
    q[base + dh]      = (_Float16)((q0 * c0 - q1 * s0) * 0.125f);
    q[base + dh + 32] = (_Float16)((q1 * c1 + q0 * s1) * 0.125f);

    float k0 = (float)k[base + dh], k1 = (float)k[base + dh + 32];
    k[base + dh]      = (_Float16)(k0 * c0 - k1 * s0);
    k[base + dh + 32] = (_Float16)(k1 * c1 + k0 * s1);
}

// ---------------------------------------------------------------------------
// Flash attention: one WG per (b, h, 64-row q tile); 4 waves, 16 rows each.
// K / V^T tiles double-buffered in LDS via async copies; online softmax with
// shfl row-reductions; P re-laid out through a per-wave LDS strip.
// ---------------------------------------------------------------------------
__global__ __launch_bounds__(128) void attn_kernel(
    const _Float16* __restrict__ q, const _Float16* __restrict__ k,
    const _Float16* __restrict__ vt, _Float16* __restrict__ attn)
{
    __shared__ __align__(16) _Float16 Ks[2][64][72];   // [key j][dh]
    __shared__ __align__(16) _Float16 Vs[2][64][72];   // [dh][key j]  (V^T)
    __shared__ __align__(16) _Float16 Ps[4][16][72];   // per-wave P strip

    const int tid  = threadIdx.x;
    const int wave = tid >> 5, lane = tid & 31;
    const int g = lane >> 4, ln = lane & 15;
    const int qt = blockIdx.x, h = blockIdx.y, b = blockIdx.z;
    const size_t bh = (size_t)b * H_ + h;
    const int qbase = qt * 64;

    // Issue one 64-key tile of K and V^T into buffer `buf` (4 async / thread).
    auto issueKV = [&](int buf, int j0) {
        int row = tid >> 1, seg = tid & 1;
        uint32_t ldsK = lds_addr(&Ks[buf][row][seg * 32]);
        uint32_t offK = (uint32_t)(((bh * S_ + j0 + row) * DH_ + seg * 32) * 2);
        ASYNC_CP16(ldsK,      offK,      k);
        ASYNC_CP16(ldsK + 16, offK + 16, k);
        uint32_t ldsV = lds_addr(&Vs[buf][row][seg * 32]);
        uint32_t offV = (uint32_t)(((bh * DH_ + row) * S_ + j0 + seg * 32) * 2);
        ASYNC_CP16(ldsV,      offV,      vt);
        ASYNC_CP16(ldsV + 16, offV + 16, vt);
    };

    // Q fragments (A layout), scale already folded in by rope_kernel
    Frag16 qa[2];
    {
        const _Float16* qrow = q + (bh * S_ + qbase + wave * 16 + ln) * DH_;
        #pragma unroll
        for (int ko2 = 0; ko2 < 2; ++ko2) {
            qa[ko2].h[0] = *(const v8h*)(qrow + ko2 * 32 + g * 8);
            qa[ko2].h[1] = *(const v8h*)(qrow + ko2 * 32 + 16 + g * 8);
        }
    }

    float mrow[8], lrow[8];
    v8f o[4];
    for (int r = 0; r < 8; ++r) { mrow[r] = -1e30f; lrow[r] = 0.0f; }
    for (int nt = 0; nt < 4; ++nt)
        for (int r = 0; r < 8; ++r) o[nt][r] = 0.0f;

    issueKV(0, 0);
    for (int j0 = 0; j0 < S_; j0 += 64) {
        const int cur = (j0 >> 6) & 1;
        if (j0 + 64 < S_) {
            issueKV(cur ^ 1, j0 + 64);   // prefetch next tile
            asm volatile("s_wait_asynccnt 0x4" ::: "memory");
        } else {
            asm volatile("s_wait_asynccnt 0x0" ::: "memory");
        }
        __syncthreads();

        // S = q * K^T : 16x64 per wave (4 n-tiles x 2 k-steps)
        v8f sc[4];
        for (int nt = 0; nt < 4; ++nt)
            for (int r = 0; r < 8; ++r) sc[nt][r] = 0.0f;
        #pragma unroll
        for (int ko2 = 0; ko2 < 2; ++ko2) {
            #pragma unroll
            for (int nt = 0; nt < 4; ++nt) {
                Frag16 bb;
                bb.h[0] = *(const v8h*)&Ks[cur][nt * 16 + ln][ko2 * 32 + g * 8];
                bb.h[1] = *(const v8h*)&Ks[cur][nt * 16 + ln][ko2 * 32 + 16 + g * 8];
                sc[nt] = __builtin_amdgcn_wmma_f32_16x16x32_f16(
                    false, qa[ko2].v, false, bb.v, (short)0, sc[nt], false, false);
            }
        }

        // Online softmax. Row m = g*8 + r spread over the 16 lanes of this half.
        float corr[8], rsum[8];
        #pragma unroll
        for (int r = 0; r < 8; ++r) {
            float tm = sc[0][r];
            #pragma unroll
            for (int nt = 1; nt < 4; ++nt) tm = fmaxf(tm, sc[nt][r]);
            tm = fmaxf(tm, __shfl_xor(tm, 1, 32));
            tm = fmaxf(tm, __shfl_xor(tm, 2, 32));
            tm = fmaxf(tm, __shfl_xor(tm, 4, 32));
            tm = fmaxf(tm, __shfl_xor(tm, 8, 32));
            float mnew = fmaxf(mrow[r], tm);
            corr[r] = __expf(mrow[r] - mnew);
            mrow[r] = mnew;
            rsum[r] = 0.0f;
        }
        #pragma unroll
        for (int nt = 0; nt < 4; ++nt) {
            #pragma unroll
            for (int r = 0; r < 8; ++r) {
                float p = __expf(sc[nt][r] - mrow[r]);
                rsum[r] += p;
                Ps[wave][g * 8 + r][nt * 16 + ln] = (_Float16)p;  // C-layout -> row-major
            }
        }
        #pragma unroll
        for (int r = 0; r < 8; ++r) {
            float rs = rsum[r];
            rs += __shfl_xor(rs, 1, 32);
            rs += __shfl_xor(rs, 2, 32);
            rs += __shfl_xor(rs, 4, 32);
            rs += __shfl_xor(rs, 8, 32);
            lrow[r] = lrow[r] * corr[r] + rs;
            #pragma unroll
            for (int nt = 0; nt < 4; ++nt) o[nt][r] *= corr[r];
        }
        __syncthreads();   // Ps strips fully written before A-frag reads

        // O += P * V^T   (B operand = Vs[n=dh][k=j], already [N][K] in LDS)
        Frag16 pa[2];
        #pragma unroll
        for (int ko2 = 0; ko2 < 2; ++ko2) {
            pa[ko2].h[0] = *(const v8h*)&Ps[wave][ln][ko2 * 32 + g * 8];
            pa[ko2].h[1] = *(const v8h*)&Ps[wave][ln][ko2 * 32 + 16 + g * 8];
        }
        #pragma unroll
        for (int ko2 = 0; ko2 < 2; ++ko2) {
            #pragma unroll
            for (int nt = 0; nt < 4; ++nt) {
                Frag16 bb;
                bb.h[0] = *(const v8h*)&Vs[cur][nt * 16 + ln][ko2 * 32 + g * 8];
                bb.h[1] = *(const v8h*)&Vs[cur][nt * 16 + ln][ko2 * 32 + 16 + g * 8];
                o[nt] = __builtin_amdgcn_wmma_f32_16x16x32_f16(
                    false, pa[ko2].v, false, bb.v, (short)0, o[nt], false, false);
            }
        }
        __syncthreads();   // done reading this buffer before it is re-issued
    }

    // attn[b, s, h*64 + dh] = O / l   (the bhid -> bs(hd) transpose)
    #pragma unroll
    for (int nt = 0; nt < 4; ++nt) {
        #pragma unroll
        for (int r = 0; r < 8; ++r) {
            int srow = qbase + wave * 16 + g * 8 + r;
            float val = o[nt][r] / lrow[r];
            attn[((size_t)b * S_ + srow) * D_ + h * DH_ + nt * 16 + ln] = (_Float16)val;
        }
    }
}

// ---------------------------------------------------------------------------
// Host launcher
// ---------------------------------------------------------------------------
extern "C" void kernel_launch(void* const* d_in, const int* in_sizes, int n_in,
                              void* d_out, int out_size, void* d_ws, size_t ws_size,
                              hipStream_t stream)
{
    const float* x    = (const float*)d_in[0];
    const float* cosp = (const float*)d_in[1];
    const float* sinp = (const float*)d_in[2];
    const float* Wqkv = (const float*)d_in[3];
    const float* Wout = (const float*)d_in[4];
    float* out = (float*)d_out;

    // Workspace layout (halves). Total live: 36M halves = 72 MB.
    _Float16* ws = (_Float16*)d_ws;
    size_t off = 0;
    _Float16* x16    = ws + off; off += (size_t)M_ * D_;    // 8M
    _Float16* wqkv16 = ws + off; off += (size_t)N3_ * D_;   // 3M
    _Float16* wout16 = ws + off; off += (size_t)D_ * D_;    // 1M
    _Float16* q16    = ws + off; off += (size_t)M_ * D_;    // 8M
    _Float16* k16    = ws + off; off += (size_t)M_ * D_;    // 8M
    _Float16* vt16   = ws + off; off += (size_t)M_ * D_;    // 8M
    _Float16* attn16 = x16;  // alias: x16 is dead after the QKV GEMM

    cvt_f32_f16<<<(M_ * D_) / 256, 256, 0, stream>>>(x, x16, M_ * D_);
    cvt_f32_f16<<<(N3_ * D_) / 256, 256, 0, stream>>>(Wqkv, wqkv16, N3_ * D_);
    cvt_f32_f16<<<(D_ * D_) / 256, 256, 0, stream>>>(Wout, wout16, D_ * D_);

    // QKV projection: [8192,1024] x [3072,1024]^T, scatter into q/k/v^T
    gemm_wmma<0><<<dim3(M_ / 128, N3_ / 64), 128, 0, stream>>>(
        x16, wqkv16, D_, N3_, q16, k16, vt16, nullptr);

    rope_kernel<<<(B_ * H_ * S_ * 32) / 256, 256, 0, stream>>>(q16, k16, cosp, sinp);

    attn_kernel<<<dim3(S_ / 64, H_, B_), 128, 0, stream>>>(q16, k16, vt16, attn16);

    // Output projection: [8192,1024] x [1024,1024]^T -> fp32 out
    gemm_wmma<1><<<dim3(M_ / 128, D_ / 64), 128, 0, stream>>>(
        attn16, wout16, D_, D_, nullptr, nullptr, nullptr, out);
}
